// BertSelfAttention_63591285784957
// MI455X (gfx1250) — matrix operand order repaired and checked by reading, not compiled, not run on hardware
//
#include <hip/hip_runtime.h>
#include <hip/hip_bf16.h>

// ---------------- constants ----------------
#define BB 8
#define SS 1024
#define DD 768
#define HH 12
#define DH 64
#define MROWS (BB*SS)                 // 8192
#define BHSD ((size_t)BB*HH*SS*DH)    // 6291456 elements per q/k/v

typedef __attribute__((ext_vector_type(16))) __bf16 v16bf;
typedef __attribute__((ext_vector_type(8)))  float  v8f;
typedef __attribute__((ext_vector_type(4)))  unsigned int u32x4;
typedef __attribute__((ext_vector_type(8)))  int i32x8;
typedef __attribute__((ext_vector_type(4)))  int i32x4;

union ABOp { v16bf v; u32x4 q[2]; unsigned int u[8]; };
union V128 { u32x4 q; unsigned short s[8]; };

__device__ __forceinline__ unsigned short f2bf(float f) {
    unsigned int u = __float_as_uint(f);
    u += 0x7FFFu + ((u >> 16) & 1u);          // round-to-nearest-even
    return (unsigned short)(u >> 16);
}

__device__ __forceinline__ unsigned lds_off(const void* p) {
    // flat LDS aperture: addr[31:0] is the wave-relative LDS byte offset
    return (unsigned)(unsigned long long)p;
}

// TDM: DMA a 2D bf16 tile (tile0 x tile1 elems, row stride stride0 elems) from
// global into LDS, padding each tile row by (pada+1) DWORDs every 2^(padi+1)
// DWORDs so the LDS stride dodges bank conflicts.  D# per ISA ch.8.
__device__ __forceinline__ void tdm_load_2d(unsigned lds_addr, const void* gptr,
                                            unsigned td0, unsigned td1,
                                            unsigned tile0, unsigned tile1,
                                            unsigned stride0,
                                            unsigned padi, unsigned pada) {
    unsigned long long ga = (unsigned long long)gptr;
    u32x4 g0;
    g0[0] = 1u;                                                  // count=1, user mode
    g0[1] = lds_addr;                                            // LDS byte address
    g0[2] = (unsigned)ga;                                        // global_addr[31:0]
    g0[3] = (unsigned)((ga >> 32) & 0x1FFFFFFu) | (2u << 30);    // addr[56:32] | type=2
    i32x8 g1;
    g1[0] = (int)((1u << 16)            // data_size = 1 -> 2 bytes
                | (1u << 20)            // pad_enable
                | (padi << 22)          // pad_interval code
                | (pada << 25));        // pad_amount code
    g1[1] = (int)((td0 & 0xFFFFu) << 16);                        // tensor_dim0 lo
    g1[2] = (int)((td0 >> 16) | ((td1 & 0xFFFFu) << 16));        // dim0 hi | dim1 lo
    g1[3] = (int)((td1 >> 16) | (tile0 << 16));                  // dim1 hi | tile_dim0
    g1[4] = (int)(tile1 & 0xFFFFu);                              // tile_dim1, tile_dim2=0
    g1[5] = (int)stride0;                                        // tensor_dim0_stride lo
    g1[6] = 0;                                                   // stride hi, dim1_stride=0
    g1[7] = 0;
    i32x4 z4 = (i32x4){0, 0, 0, 0};
    i32x8 z8 = (i32x8){0, 0, 0, 0, 0, 0, 0, 0};
    // clang-23 / therock-10.0 6-arg form (groups 2..4 disabled/zero, cpol=0)
    __builtin_amdgcn_tensor_load_to_lds(g0, g1, z4, z4, z8, 0);
}

// ---------------- fp32 -> bf16 convert ----------------
__global__ void cvt_bf16(const float* __restrict__ src,
                         unsigned short* __restrict__ dst, int n) {
    int i = blockIdx.x * blockDim.x + threadIdx.x;
    if (i < n) dst[i] = f2bf(src[i]);
}

// ---------------- QKV projection GEMM ----------------
// out[m,n] = sum_d hs[m,d] * W[n,d] + bias[n];  q scaled by 0.125
// grid: (M/64, N/64, 3 matrices), block 128 (4 waves), wave w -> 16x64 strip
#define GST 40   // LDS row stride (shorts): 32 data + 8 pad, 80B (16B aligned)
__global__ __launch_bounds__(128) void qkv_gemm(
    const unsigned short* __restrict__ hsb,   // [8192][768] bf16
    const unsigned short* __restrict__ Wb,    // [3][768][768] bf16
    const float* __restrict__ bq, const float* __restrict__ bk,
    const float* __restrict__ bvv,
    unsigned short* __restrict__ qkvb)        // [3][B*H][S][64] bf16
{
    __shared__ unsigned short At[64 * GST];
    __shared__ unsigned short Bt[64 * GST];

    const int tid = threadIdx.x, lane = tid & 31, w = tid >> 5;
    const int nq = lane & 15, hh = lane >> 4;
    const int mat = blockIdx.z;
    const int m0 = blockIdx.x * 64;
    const int n0 = blockIdx.y * 64;
    const unsigned short* Wm = Wb + (size_t)mat * DD * DD;
    const unsigned at_lds = lds_off(At), bt_lds = lds_off(Bt);

    v8f acc[4];
    #pragma unroll
    for (int i = 0; i < 4; ++i) acc[i] = (v8f){0.f,0.f,0.f,0.f,0.f,0.f,0.f,0.f};

    const int am = w * 16 + nq;

    for (int k0 = 0; k0 < DD; k0 += 32) {
        __syncthreads();                       // previous tile fully consumed
        if (w == 0) {                          // TDM DMA: A tile 32x64, B tile 32x64
            tdm_load_2d(at_lds, hsb + (size_t)m0 * DD + k0, DD, MROWS, 32, 64, DD,
                        /*padi 16DW*/3, /*pada 4DW*/3);
            tdm_load_2d(bt_lds, Wm + (size_t)n0 * DD + k0, DD, DD, 32, 64, DD, 3, 3);
            __builtin_amdgcn_s_wait_tensorcnt(0);
        }
        __syncthreads();

        ABOp a;                                // A-layout: K = 16(v/4)+8h+2(v%4)
        a.q[0] = *(const u32x4*)(&At[am * GST + 8 * hh]);
        a.q[1] = *(const u32x4*)(&At[am * GST + 16 + 8 * hh]);
        #pragma unroll
        for (int ns = 0; ns < 4; ++ns) {
            ABOp bo;                           // B-layout: K = 16(v/4)+8h+2(v%4)
            const int bn = ns * 16 + nq;
            bo.q[0] = *(const u32x4*)(&Bt[bn * GST + 8 * hh]);
            bo.q[1] = *(const u32x4*)(&Bt[bn * GST + 16 + 8 * hh]);
            acc[ns] = __builtin_amdgcn_wmma_f32_16x16x32_bf16(
                false, a.v, false, bo.v, (short)0, acc[ns], false, false);
        }
    }

    const float* bias = (mat == 0) ? bq : ((mat == 1) ? bk : bvv);
    const float scale = (mat == 0) ? 0.125f : 1.0f;   // fold 1/sqrt(64) into Q
    #pragma unroll
    for (int ns = 0; ns < 4; ++ns) {
        const int ng = n0 + ns * 16 + nq;
        const float bsv = bias[ng];
        const int head = ng >> 6, dd = ng & 63;
        #pragma unroll
        for (int r = 0; r < 8; ++r) {
            int mg = m0 + w * 16 + r + 8 * hh;         // C/D row layout
            int b = mg >> 10, s = mg & 1023;
            float val = (acc[ns][r] + bsv) * scale;
            qkvb[(size_t)mat * BHSD + (((size_t)(b * HH + head)) * SS + s) * DH + dd] = f2bf(val);
        }
    }
}

// ---------------- flash attention ----------------
// grid: (S/64, B*H), block 128 (4 waves); wave w: q rows [blk*64+w*16, +16)
#define KST 72   // 64 data + 8 pad shorts, 144B rows (16B aligned)
#define VST 40   // 32 data + 8 pad shorts, 80B rows (16B aligned)
__global__ __launch_bounds__(128) void attn(
    const unsigned short* __restrict__ qkvb,
    const float* __restrict__ mask,    // [B,1,1,S] -> b*S+s
    const float* __restrict__ click,   // [B,S]
    float* __restrict__ out)           // [B,S,768] f32
{
    __shared__ unsigned short Kt[32 * KST];
    __shared__ unsigned short Vt[64 * VST];    // transposed: Vt[d][key]
    __shared__ unsigned short Pb[4 * 16 * 32];

    const int tid = threadIdx.x, lane = tid & 31, w = tid >> 5;
    const int nq = lane & 15, hh = lane >> 4;
    const int bh = blockIdx.y;
    const int b = bh / HH, hd = bh % HH;
    const unsigned short* qp = qkvb + (size_t)bh * SS * DH;
    const unsigned short* kp = qkvb + BHSD + (size_t)bh * SS * DH;
    const unsigned short* vp = qkvb + 2 * BHSD + (size_t)bh * SS * DH;
    const int q0 = blockIdx.x * 64 + w * 16;
    const unsigned kt_lds = lds_off(Kt);

    // Q operand registers: 16x64 = two 16x32 A operands (scale pre-folded), b128 loads
    ABOp qa[2];
    #pragma unroll
    for (int c = 0; c < 2; ++c) {
        qa[c].q[0] = *(const u32x4*)(qp + (size_t)(q0 + nq) * DH + c * 32 + 8 * hh);
        qa[c].q[1] = *(const u32x4*)(qp + (size_t)(q0 + nq) * DH + c * 32 + 16 + 8 * hh);
    }

    float mi[8], li[8];
    v8f acc[4];
    #pragma unroll
    for (int r = 0; r < 8; ++r) { mi[r] = -1e30f; li[r] = 0.f; }
    #pragma unroll
    for (int dt = 0; dt < 4; ++dt) acc[dt] = (v8f){0.f,0.f,0.f,0.f,0.f,0.f,0.f,0.f};

    unsigned short* Pw = Pb + w * 16 * 32;
    const int vr = tid >> 2;                // V staging: key row 0..31
    const int vc = (tid & 3) * 16;          // d base (16 shorts per thread)

    for (int it = 0; it < SS / 32; ++it) {
        const int kt0 = it * 32;
        __syncthreads();                    // previous tile fully consumed

        // V gather: two b128 global loads (overlaps K-tile TDM below)
        V128 v0, v1;
        v0.q = *(const u32x4*)(vp + (size_t)(kt0 + vr) * DH + vc);
        v1.q = *(const u32x4*)(vp + (size_t)(kt0 + vr) * DH + vc + 8);

        if (w == 0)                         // K tile 64x32 via Tensor Data Mover
            tdm_load_2d(kt_lds, kp + (size_t)kt0 * DH, DH, SS, DH, 32, DH,
                        /*padi 32DW*/4, /*pada 4DW*/3);

        if (it + 1 < SS / 32)               // prefetch next V tile
            __builtin_prefetch(vp + (size_t)(kt0 + 32) * DH + tid * 16, 0, 3);

        // V scatter transposed into LDS
        #pragma unroll
        for (int j = 0; j < 8; ++j) {
            Vt[(vc + j) * VST + vr]     = v0.s[j];
            Vt[(vc + 8 + j) * VST + vr] = v1.s[j];
        }
        if (w == 0) __builtin_amdgcn_s_wait_tensorcnt(0);
        __syncthreads();

        // scores: two 16(q)x16(key) tiles, K-dim = DH in 2 chunks
        v8f sc[2];
        #pragma unroll
        for (int t = 0; t < 2; ++t) {
            v8f s = (v8f){0.f,0.f,0.f,0.f,0.f,0.f,0.f,0.f};
            const int key = t * 16 + nq;
            #pragma unroll
            for (int c = 0; c < 2; ++c) {
                ABOp bo;                    // B-layout: K(d) = c*32 + 16(v/4) + 8h + 2(v%4)
                bo.q[0] = *(const u32x4*)(&Kt[key * KST + c * 32 + 8 * hh]);
                bo.q[1] = *(const u32x4*)(&Kt[key * KST + c * 32 + 16 + 8 * hh]);
                s = __builtin_amdgcn_wmma_f32_16x16x32_bf16(
                    false, qa[c].v, false, bo.v, (short)0, s, false, false);
            }
            sc[t] = s;
        }

        // multiplicative click gate, then additive mask (reference order)
        const float ct0 = click[b * SS + kt0 + nq];
        const float mk0 = mask [b * SS + kt0 + nq];
        const float ct1 = click[b * SS + kt0 + 16 + nq];
        const float mk1 = mask [b * SS + kt0 + 16 + nq];

        #pragma unroll
        for (int r = 0; r < 8; ++r) {
            float s0 = sc[0][r] * ct0 + mk0;
            float s1 = sc[1][r] * ct1 + mk1;
            float mx = fmaxf(s0, s1);
            #pragma unroll
            for (int off = 8; off >= 1; off >>= 1)
                mx = fmaxf(mx, __shfl_xor(mx, off, 32));   // within 16-lane half
            float mnew = fmaxf(mi[r], mx);
            float rs = __expf(mi[r] - mnew);
            float p0 = __expf(s0 - mnew);
            float p1 = __expf(s1 - mnew);
            float sum = p0 + p1;
            #pragma unroll
            for (int off = 8; off >= 1; off >>= 1)
                sum += __shfl_xor(sum, off, 32);
            li[r] = li[r] * rs + sum;
            mi[r] = mnew;
            #pragma unroll
            for (int dt = 0; dt < 4; ++dt) acc[dt][r] *= rs;
            const int m = r + 8 * hh;                      // D-layout row
            Pw[m * 32 + nq]      = f2bf(p0);
            Pw[m * 32 + 16 + nq] = f2bf(p1);
        }
        asm volatile("s_wait_dscnt 0x0" ::: "memory");     // intra-wave LDS RAW on P

        // ctx += P(16x32) x V(32x64)
        ABOp pa;                                           // A-layout from P buffer
        pa.q[0] = *(const u32x4*)(&Pw[nq * 32 + 8 * hh]);
        pa.q[1] = *(const u32x4*)(&Pw[nq * 32 + 16 + 8 * hh]);
        #pragma unroll
        for (int dt = 0; dt < 4; ++dt) {
            ABOp bo;                                       // B-layout: K(key) = 16(v/4)+8h+2(v%4)
            const int dcol = dt * 16 + nq;
            bo.q[0] = *(const u32x4*)(&Vt[dcol * VST + 8 * hh]);
            bo.q[1] = *(const u32x4*)(&Vt[dcol * VST + 16 + 8 * hh]);
            acc[dt] = __builtin_amdgcn_wmma_f32_16x16x32_bf16(
                false, pa.v, false, bo.v, (short)0, acc[dt], false, false);
        }
    }

    // epilogue: normalize and store f32 [B,S,768]
    #pragma unroll
    for (int dt = 0; dt < 4; ++dt) {
        const int dcol = hd * DH + dt * 16 + nq;
        #pragma unroll
        for (int r = 0; r < 8; ++r) {
            int m = q0 + r + 8 * hh;
            out[((size_t)b * SS + m) * DD + dcol] = acc[dt][r] / li[r];
        }
    }
}

// ---------------- launcher ----------------
extern "C" void kernel_launch(void* const* d_in, const int* in_sizes, int n_in,
                              void* d_out, int out_size, void* d_ws, size_t ws_size,
                              hipStream_t stream) {
    const float* hs    = (const float*)d_in[0];
    const float* amask = (const float*)d_in[1];
    const float* click = (const float*)d_in[2];
    const float* Wq    = (const float*)d_in[3];
    const float* bq    = (const float*)d_in[4];
    const float* Wk    = (const float*)d_in[5];
    const float* bk    = (const float*)d_in[6];
    const float* Wv    = (const float*)d_in[7];
    const float* bv    = (const float*)d_in[8];
    float* out = (float*)d_out;

    unsigned short* ws = (unsigned short*)d_ws;
    unsigned short* hsb  = ws;                                   // 8192*768
    unsigned short* Wb   = hsb + (size_t)MROWS * DD;             // 3*768*768
    unsigned short* qkvb = Wb + (size_t)3 * DD * DD;             // 3*BHSD

    const int nhs = MROWS * DD;          // 6291456
    const int nw  = DD * DD;             // 589824
    cvt_bf16<<<(nhs + 255) / 256, 256, 0, stream>>>(hs, hsb, nhs);
    cvt_bf16<<<(nw + 255) / 256, 256, 0, stream>>>(Wq, Wb + 0 * (size_t)nw, nw);
    cvt_bf16<<<(nw + 255) / 256, 256, 0, stream>>>(Wk, Wb + 1 * (size_t)nw, nw);
    cvt_bf16<<<(nw + 255) / 256, 256, 0, stream>>>(Wv, Wb + 2 * (size_t)nw, nw);

    qkv_gemm<<<dim3(MROWS / 64, DD / 64, 3), 128, 0, stream>>>(
        hsb, Wb, bq, bk, bv, qkvb);

    attn<<<dim3(SS / 64, BB * HH), 128, 0, stream>>>(qkvb, amask, click, out);
}
